// MoE_64372969832825
// MI455X (gfx1250) — matrix operand order repaired and checked by reading, not compiled, not run on hardware
//
#include <hip/hip_runtime.h>
#include <hip/hip_bf16.h>
#include <math.h>

// Problem constants (from reference): B=2,S=2048,D=2048,H=1024,E=16,K=2
#define T_TOK 4096
#define DDIM  2048
#define HDIM  1024
#define NEXP  16
#define CAPE  512   // capacity per expert = K*T/E

typedef __attribute__((ext_vector_type(16))) __bf16 v16bf;
typedef __attribute__((ext_vector_type(8)))  float  v8f;
typedef __attribute__((ext_vector_type(2)))  __bf16 v2bf;
typedef __attribute__((ext_vector_type(2)))  float  v2f;

// Fragment viewed either as 2x uint4 (for b128 LDS loads) or 16x bf16 (for WMMA)
union FragU { uint4 q[2]; v16bf v; };

// pack two fp32 -> two bf16 (RNE) in one dword: low half = first (even K).
// Vector conversion lowers to a single v_cvt_pk_bf16_f32 when selectable.
static __device__ __forceinline__ unsigned pk2(float a, float b) {
#if __has_builtin(__builtin_amdgcn_cvt_pk_bf16_f32)
  union { v2bf h; unsigned u; } v;
  v.h = __builtin_amdgcn_cvt_pk_bf16_f32(a, b);
  return v.u;
#else
  v2f f; f.x = a; f.y = b;
  union { v2bf h; unsigned u; } v;
  v.h = __builtin_convertvector(f, v2bf);
  return v.u;
#endif
}

// ---------------------------------------------------------------------------
// Kernel 1: per-token gating in exact fp32 (one wave32 per token).
// ---------------------------------------------------------------------------
__global__ __launch_bounds__(32) void gate_kernel(
    const float* __restrict__ x, const float* __restrict__ gw,
    const float* __restrict__ cw, const float* __restrict__ cb,
    float* __restrict__ probs, int* __restrict__ tidx,
    float* __restrict__ tgate, float* __restrict__ coef)
{
  const int t    = blockIdx.x;
  const int lane = threadIdx.x;
  const float* xr = x + (size_t)t * DDIM;

  float logit[NEXP];
  #pragma unroll
  for (int e = 0; e < NEXP; ++e) {
    float s = 0.f;
    for (int d = lane; d < DDIM; d += 32) s += xr[d] * gw[(size_t)d * NEXP + e];
    #pragma unroll
    for (int o = 16; o > 0; o >>= 1) s += __shfl_xor(s, o, 32);
    logit[e] = s;
  }
  float c0 = 0.f, c1 = 0.f;
  for (int d = lane; d < DDIM; d += 32) {
    float xv = xr[d];
    c0 += xv * cw[(size_t)d * 2 + 0];
    c1 += xv * cw[(size_t)d * 2 + 1];
  }
  #pragma unroll
  for (int o = 16; o > 0; o >>= 1) { c0 += __shfl_xor(c0, o, 32); c1 += __shfl_xor(c1, o, 32); }
  c0 += cb[0]; c1 += cb[1];

  float mx = logit[0];
  #pragma unroll
  for (int e = 1; e < NEXP; ++e) mx = fmaxf(mx, logit[e]);
  float p[NEXP], ssum = 0.f;
  #pragma unroll
  for (int e = 0; e < NEXP; ++e) { p[e] = expf(logit[e] - mx); ssum += p[e]; }
  float inv = 1.f / ssum;
  #pragma unroll
  for (int e = 0; e < NEXP; ++e) p[e] *= inv;

  int   i0 = 0; float v0 = p[0];
  #pragma unroll
  for (int e = 1; e < NEXP; ++e) if (p[e] > v0) { v0 = p[e]; i0 = e; }
  int   i1 = -1; float v1 = -1.f;
  #pragma unroll
  for (int e = 0; e < NEXP; ++e) if (e != i0 && p[e] > v1) { v1 = p[e]; i1 = e; }
  float gs = v0 + v1;
  float g0 = v0 / gs, g1 = v1 / gs;

  float cm = fmaxf(c0, c1);
  float e0 = expf(c0 - cm), e1 = expf(c1 - cm);
  float ci = 1.f / (e0 + e1);

  if (lane == 0) {
    #pragma unroll
    for (int e = 0; e < NEXP; ++e) probs[(size_t)t * NEXP + e] = p[e];
    tidx [t * 2 + 0] = i0;  tidx [t * 2 + 1] = i1;
    tgate[t * 2 + 0] = g0;  tgate[t * 2 + 1] = g1;
    coef [t * 2 + 0] = e0 * ci;  coef[t * 2 + 1] = e1 * ci;
  }
}

// ---------------------------------------------------------------------------
// Kernel 2: GShard capacity scan (deterministic serial per-expert scan).
// ---------------------------------------------------------------------------
__global__ __launch_bounds__(32) void scan_kernel(
    const float* __restrict__ probs, const int* __restrict__ tidx,
    const float* __restrict__ tgate,
    int* __restrict__ slot_token, float* __restrict__ slot_gate,
    float* __restrict__ out_tail)
{
  __shared__ float red[NEXP];
  const int e = threadIdx.x;
  if (e < NEXP) {
    for (int c = 0; c < CAPE; ++c) { slot_token[e * CAPE + c] = -1; slot_gate[e * CAPE + c] = 0.f; }
    float me = 0.f, ce = 0.f;
    int cnt = 0;
    for (int t = 0; t < T_TOK; ++t) me += probs[(size_t)t * NEXP + e];
    for (int t = 0; t < T_TOK; ++t) {                 // k = 0 pass
      if (tidx[t * 2] == e) {
        if (cnt < CAPE) { slot_token[e * CAPE + cnt] = t; slot_gate[e * CAPE + cnt] = tgate[t * 2]; }
        ++cnt; ce += 1.f;
      }
    }
    for (int t = 0; t < T_TOK; ++t) {                 // k = 1 pass (prior = k0 count)
      if (tidx[t * 2 + 1] == e) {
        if (cnt < CAPE) { slot_token[e * CAPE + cnt] = t; slot_gate[e * CAPE + cnt] = tgate[t * 2 + 1]; }
        ++cnt;
      }
    }
    out_tail[1 + e] = (float)cnt;                     // exp_counts (uncapped)
    red[e] = (me / (float)T_TOK) * (ce / (float)T_TOK);
  }
  __syncthreads();
  if (e == 0) {
    float s = 0.f;
    for (int i = 0; i < NEXP; ++i) s += red[i];
    out_tail[0] = s * (float)NEXP;                    // l_aux
  }
}

// ---------------------------------------------------------------------------
// LDS layout (fragment-order, packed bf16 pairs, ping-pong buffered):
//   lA[p][row][kp] / lB[p][n][kp]; rows = 16 k-pair dwords + 4 pad = 80B,
//   16B aligned. Fragment loads = two 16B ds_load_b128 per operand.
// One barrier per K-tile: store(i+1 -> buf 1-p) overlaps compute(i, buf p).
// ---------------------------------------------------------------------------

// Kernel 3: fused SwiGLU up-proj: Hout = silu(A@W1) * (A@W3).
__global__ __launch_bounds__(256) void gemm1_kernel(
    const float* __restrict__ A, const int* __restrict__ row_map,
    const float* __restrict__ W1, const float* __restrict__ W3,
    size_t wstride, float* __restrict__ Hout, int N, int Kdim)
{
  __shared__ unsigned lA [2][128][20];
  __shared__ unsigned lB1[2][ 64][20];
  __shared__ unsigned lB3[2][ 64][20];

  const int tid  = threadIdx.x;
  const int lane = tid & 31;
  const int wave = tid >> 5;
  const int rowBase = blockIdx.x * 128;
  const int colBase = blockIdx.y * 64;
  const int eidx = rowBase / CAPE;
  const float* w1p = W1 + (size_t)eidx * wstride;
  const float* w3p = W3 + (size_t)eidx * wstride;

  // A staging: thread -> (row 0..127, k-half 0/16)
  const int ar = tid >> 1;
  const int au = (tid & 1) * 8;
  const int grow = row_map ? row_map[rowBase + ar] : (rowBase + ar);
  const bool avalid = (grow >= 0);
  const float4* ap4 = avalid ? (const float4*)(A + (size_t)grow * Kdim + (tid & 1) * 16)
                             : nullptr;
  // B staging: thread -> (k-pair 0..15, 4 n's)
  const int kp = tid >> 4;
  const int nq = (tid & 15) * 4;

  v8f acc1[4], acc3[4];
  #pragma unroll
  for (int i = 0; i < 4; ++i) {
    #pragma unroll
    for (int j = 0; j < 8; ++j) { acc1[i][j] = 0.f; acc3[i][j] = 0.f; }
  }

  const int mr  = wave * 16 + (lane & 15);
  const int kuA = (lane < 16) ? 0 : 8;
  const int kuB = (lane < 16) ? 0 : 8;
  const int nc  = lane & 15;

  float4 a0, a1, a2, a3, r1a, r1b, r3a, r3b;
  a0 = a1 = a2 = a3 = make_float4(0.f, 0.f, 0.f, 0.f);

  auto loadT = [&](int t) {
    const int kk = t * 32;
    if (avalid) {
      a0 = ap4[kk / 4 + 0]; a1 = ap4[kk / 4 + 1];
      a2 = ap4[kk / 4 + 2]; a3 = ap4[kk / 4 + 3];
    }
    const float* b1r = w1p + (size_t)(kk + 2 * kp) * N + colBase + nq;
    const float* b3r = w3p + (size_t)(kk + 2 * kp) * N + colBase + nq;
    r1a = *(const float4*)b1r;  r1b = *(const float4*)(b1r + N);
    r3a = *(const float4*)b3r;  r3b = *(const float4*)(b3r + N);
  };
  auto storeT = [&](int pb) {
    uint4 u0 = make_uint4(pk2(a0.x, a0.y), pk2(a0.z, a0.w),
                          pk2(a1.x, a1.y), pk2(a1.z, a1.w));
    uint4 u1 = make_uint4(pk2(a2.x, a2.y), pk2(a2.z, a2.w),
                          pk2(a3.x, a3.y), pk2(a3.z, a3.w));
    *(uint4*)&lA[pb][ar][au]     = u0;
    *(uint4*)&lA[pb][ar][au + 4] = u1;
    lB1[pb][nq + 0][kp] = pk2(r1a.x, r1b.x);
    lB1[pb][nq + 1][kp] = pk2(r1a.y, r1b.y);
    lB1[pb][nq + 2][kp] = pk2(r1a.z, r1b.z);
    lB1[pb][nq + 3][kp] = pk2(r1a.w, r1b.w);
    lB3[pb][nq + 0][kp] = pk2(r3a.x, r3b.x);
    lB3[pb][nq + 1][kp] = pk2(r3a.y, r3b.y);
    lB3[pb][nq + 2][kp] = pk2(r3a.z, r3b.z);
    lB3[pb][nq + 3][kp] = pk2(r3a.w, r3b.w);
  };

  const int NT = Kdim / 32;
  loadT(0);
  storeT(0);
  if (NT > 1) loadT(1);
  __syncthreads();

  int p = 0;
  for (int i = 0; i < NT; ++i) {
    if (i + 1 < NT) {
      storeT(1 - p);                     // stage tile i+1 into the other buffer
      if (i + 2 < NT) loadT(i + 2);      // global prefetch overlaps compute
    }
    // compute tile i from buf p
    FragU fa;
    fa.q[0] = *(const uint4*)&lA[p][mr][kuA];
    fa.q[1] = *(const uint4*)&lA[p][mr][kuA + 4];
    #pragma unroll
    for (int s = 0; s < 4; ++s) {
      const int nn = s * 16 + nc;
      FragU fb1, fb3;
      fb1.q[0] = *(const uint4*)&lB1[p][nn][kuB];
      fb1.q[1] = *(const uint4*)&lB1[p][nn][kuB + 4];
      fb3.q[0] = *(const uint4*)&lB3[p][nn][kuB];
      fb3.q[1] = *(const uint4*)&lB3[p][nn][kuB + 4];
      acc1[s] = __builtin_amdgcn_wmma_f32_16x16x32_bf16(false, fa.v, false, fb1.v,
                                                        (short)0, acc1[s], false, false);
      acc3[s] = __builtin_amdgcn_wmma_f32_16x16x32_bf16(false, fa.v, false, fb3.v,
                                                        (short)0, acc3[s], false, false);
    }
    __syncthreads();
    p ^= 1;
  }

  const int rAdd = (lane < 16) ? 0 : 8;               // C/D: vgpr r -> M=r / r+8
  #pragma unroll
  for (int s = 0; s < 4; ++s) {
    #pragma unroll
    for (int r = 0; r < 8; ++r) {
      float v1 = acc1[s][r], v3 = acc3[s][r];
      float hv = (v1 / (1.f + __expf(-v1))) * v3;     // silu(v1) * v3
      int orow = rowBase + wave * 16 + r + rAdd;
      int ocol = colBase + s * 16 + nc;
      Hout[(size_t)orow * N + ocol] = hv;
    }
  }
}

// ---------------------------------------------------------------------------
// Kernel 4: down-proj GEMM with fused combine.
//   mode 0 (shared):  Out[row, col]  = coef1[row] * (A@W)
//   mode 1 (expert):  tok = slot_token[row]; if tok>=0:
//                     Out[tok, col] += coef0[tok]*slot_gate[row]*(A@W)  (atomic)
// ---------------------------------------------------------------------------
__global__ __launch_bounds__(256) void gemm2_kernel(
    const float* __restrict__ A, const float* __restrict__ W, size_t wstride,
    float* __restrict__ Out, int N, int Kdim, int mode,
    const int* __restrict__ slot_token, const float* __restrict__ slot_gate,
    const float* __restrict__ coef)
{
  __shared__ unsigned lA[2][128][20];
  __shared__ unsigned lB[2][ 64][20];

  const int tid  = threadIdx.x;
  const int lane = tid & 31;
  const int wave = tid >> 5;
  const int rowBase = blockIdx.x * 128;
  const int colBase = blockIdx.y * 64;
  const int eidx = rowBase / CAPE;
  const float* wp = W + (size_t)eidx * wstride;

  const int ar = tid >> 1;
  const int au = (tid & 1) * 8;
  const float4* ap4 = (const float4*)(A + (size_t)(rowBase + ar) * Kdim + (tid & 1) * 16);
  const int kp = tid >> 4;
  const int nq = (tid & 15) * 4;

  v8f acc[4];
  #pragma unroll
  for (int i = 0; i < 4; ++i) {
    #pragma unroll
    for (int j = 0; j < 8; ++j) acc[i][j] = 0.f;
  }

  const int mr  = wave * 16 + (lane & 15);
  const int kuA = (lane < 16) ? 0 : 8;
  const int kuB = (lane < 16) ? 0 : 8;
  const int nc  = lane & 15;

  float4 a0, a1, a2, a3, rba, rbb;

  auto loadT = [&](int t) {
    const int kk = t * 32;
    a0 = ap4[kk / 4 + 0]; a1 = ap4[kk / 4 + 1];
    a2 = ap4[kk / 4 + 2]; a3 = ap4[kk / 4 + 3];
    const float* br = wp + (size_t)(kk + 2 * kp) * N + colBase + nq;
    rba = *(const float4*)br;  rbb = *(const float4*)(br + N);
  };
  auto storeT = [&](int pb) {
    uint4 u0 = make_uint4(pk2(a0.x, a0.y), pk2(a0.z, a0.w),
                          pk2(a1.x, a1.y), pk2(a1.z, a1.w));
    uint4 u1 = make_uint4(pk2(a2.x, a2.y), pk2(a2.z, a2.w),
                          pk2(a3.x, a3.y), pk2(a3.z, a3.w));
    *(uint4*)&lA[pb][ar][au]     = u0;
    *(uint4*)&lA[pb][ar][au + 4] = u1;
    lB[pb][nq + 0][kp] = pk2(rba.x, rbb.x);
    lB[pb][nq + 1][kp] = pk2(rba.y, rbb.y);
    lB[pb][nq + 2][kp] = pk2(rba.z, rbb.z);
    lB[pb][nq + 3][kp] = pk2(rba.w, rbb.w);
  };

  const int NT = Kdim / 32;
  loadT(0);
  storeT(0);
  if (NT > 1) loadT(1);
  __syncthreads();

  int p = 0;
  for (int i = 0; i < NT; ++i) {
    if (i + 1 < NT) {
      storeT(1 - p);
      if (i + 2 < NT) loadT(i + 2);
    }
    FragU fa;
    fa.q[0] = *(const uint4*)&lA[p][mr][kuA];
    fa.q[1] = *(const uint4*)&lA[p][mr][kuA + 4];
    #pragma unroll
    for (int s = 0; s < 4; ++s) {
      const int nn = s * 16 + nc;
      FragU fb;
      fb.q[0] = *(const uint4*)&lB[p][nn][kuB];
      fb.q[1] = *(const uint4*)&lB[p][nn][kuB + 4];
      acc[s] = __builtin_amdgcn_wmma_f32_16x16x32_bf16(false, fa.v, false, fb.v,
                                                       (short)0, acc[s], false, false);
    }
    __syncthreads();
    p ^= 1;
  }

  const int rAdd = (lane < 16) ? 0 : 8;
  #pragma unroll
  for (int s = 0; s < 4; ++s) {
    #pragma unroll
    for (int r = 0; r < 8; ++r) {
      float v   = acc[s][r];
      int  row  = rowBase + wave * 16 + r + rAdd;
      int  col  = colBase + s * 16 + nc;
      if (mode == 0) {
        Out[(size_t)row * N + col] = coef[row * 2 + 1] * v;
      } else {
        int tok = slot_token[row];
        if (tok >= 0)
          atomicAdd(&Out[(size_t)tok * N + col], coef[tok * 2] * slot_gate[row] * v);
      }
    }
  }
}

// ---------------------------------------------------------------------------
extern "C" void kernel_launch(void* const* d_in, const int* in_sizes, int n_in,
                              void* d_out, int out_size, void* d_ws, size_t ws_size,
                              hipStream_t stream) {
  const float* x   = (const float*)d_in[0];  // [T, D]
  const float* gw  = (const float*)d_in[1];  // [D, E]
  const float* w1  = (const float*)d_in[2];  // [E, D, H]
  const float* w3  = (const float*)d_in[3];  // [E, D, H]
  const float* w2  = (const float*)d_in[4];  // [E, H, D]
  const float* sw1 = (const float*)d_in[5];  // [D, H]
  const float* sw3 = (const float*)d_in[6];  // [D, H]
  const float* sw2 = (const float*)d_in[7];  // [H, D]
  const float* cw  = (const float*)d_in[8];  // [D, 2]
  const float* cb  = (const float*)d_in[9];  // [2]
  float* out = (float*)d_out;                // [T*D] out, [1] l_aux, [E] counts

  char* ws = (char*)d_ws;
  float* probs      = (float*)ws;  ws += (size_t)T_TOK * NEXP * 4;
  int*   tidx       = (int*)  ws;  ws += (size_t)T_TOK * 2 * 4;
  float* tgate      = (float*)ws;  ws += (size_t)T_TOK * 2 * 4;
  float* coef       = (float*)ws;  ws += (size_t)T_TOK * 2 * 4;
  int*   slot_token = (int*)  ws;  ws += (size_t)NEXP * CAPE * 4;
  float* slot_gate  = (float*)ws;  ws += (size_t)NEXP * CAPE * 4;
  float* hs_shared  = (float*)ws;  ws += (size_t)T_TOK * HDIM * 4;        // 16 MB
  float* hs_expert  = (float*)ws;                                         // 32 MB

  const int Mexp = NEXP * CAPE;  // 8192 dispatched rows

  gate_kernel<<<T_TOK, 32, 0, stream>>>(x, gw, cw, cb, probs, tidx, tgate, coef);
  scan_kernel<<<1, 32, 0, stream>>>(probs, tidx, tgate, slot_token, slot_gate,
                                    out + (size_t)T_TOK * DDIM);
  gemm1_kernel<<<dim3(T_TOK / 128, HDIM / 64), 256, 0, stream>>>(
      x, nullptr, sw1, sw3, (size_t)0, hs_shared, HDIM, DDIM);
  gemm1_kernel<<<dim3(Mexp / 128, HDIM / 64), 256, 0, stream>>>(
      x, slot_token, w1, w3, (size_t)DDIM * HDIM, hs_expert, HDIM, DDIM);
  gemm2_kernel<<<dim3(T_TOK / 128, DDIM / 64), 256, 0, stream>>>(
      hs_shared, sw2, (size_t)0, out, DDIM, HDIM, 0, nullptr, nullptr, coef);
  gemm2_kernel<<<dim3(Mexp / 128, DDIM / 64), 256, 0, stream>>>(
      hs_expert, w2, (size_t)HDIM * DDIM, out, DDIM, HDIM, 1, slot_token, slot_gate, coef);
}
